// HierarchicalResonanceLayer_16286515986761
// MI455X (gfx1250) — matrix operand order repaired and checked by reading, not compiled, not run on hardware
//
#include <hip/hip_runtime.h>
#include <hip/hip_bf16.h>
#include <math.h>

typedef _Float16 v16h __attribute__((ext_vector_type(16)));
typedef _Float16 v8h  __attribute__((ext_vector_type(8)));
typedef _Float16 v4h  __attribute__((ext_vector_type(4)));
typedef float    v8f  __attribute__((ext_vector_type(8)));
typedef int      v4i  __attribute__((ext_vector_type(4)));

// address-space-qualified element types for the async builtin
typedef __attribute__((address_space(1))) v4i glob_v4i;
typedef __attribute__((address_space(3))) v4i lds_v4i;

#define B_ROWS   32768
#define KDIM     2048
#define ND       32
#define NTH      128
#define NG       512
#define NTOT     672            // 32 + 128 + 512
#define WPB      8              // waves per block (wave32)

// GEMM block tiling
#define MB   128                // block M tile
#define NB   96                 // block N tile (672 = 7 * 96)
#define KC   64                 // K stage
#define LDSS 72                 // padded LDS row stride in halves (144B -> conflict-free b128)

#define TWO_PI_F 6.28318530717958647692f
#define EPS_F    1e-6f
#define DT_F     0.01f
#define KC_F     2.0f
#define F_DELTA_F 2.0f
#define F_THETA_F 6.0f
#define F_GAMMA_F 40.0f

// ---- CDNA5 async global->LDS path (guarded; falls back to sync copy) ----
#if defined(__has_builtin)
#if __has_builtin(__builtin_amdgcn_global_load_async_to_lds_b128)
#define HRL_HAS_ASYNC 1
#endif
#endif
#ifndef HRL_HAS_ASYNC
#define HRL_HAS_ASYNC 0
#endif

__device__ __forceinline__ void hrl_wait_async() {
#if HRL_HAS_ASYNC
#if __has_builtin(__builtin_amdgcn_s_wait_asynccnt)
    __builtin_amdgcn_s_wait_asynccnt(0);
#else
    asm volatile("s_wait_asynccnt 0" ::: "memory");
#endif
#endif
}

// ---------------- conversion: x (f32) -> f16 in workspace ----------------
__global__ void hrl_cvt_x_f16(const float* __restrict__ x, _Float16* __restrict__ o, int n4) {
    int i = blockIdx.x * blockDim.x + threadIdx.x;
    const int stride = gridDim.x * blockDim.x;
    for (; i < n4; i += stride) {
        const float4 f = ((const float4*)x)[i];
        v4h h = { (_Float16)f.x, (_Float16)f.y, (_Float16)f.z, (_Float16)f.w };
        ((v4h*)o)[i] = h;
    }
}

// -------- pack W_delta|W_theta|W_gamma into one f16 B-matrix (672 x 2048) --------
__global__ void hrl_cvt_w_f16(const float* __restrict__ wd, const float* __restrict__ wt,
                              const float* __restrict__ wg, _Float16* __restrict__ o, int n4) {
    int i = blockIdx.x * blockDim.x + threadIdx.x;
    if (i >= n4) return;
    const int row = i / (KDIM / 4);
    const int k4  = i - row * (KDIM / 4);
    const float* src = (row < ND)       ? wd + (size_t)row * KDIM
                     : (row < ND + NTH) ? wt + (size_t)(row - ND) * KDIM
                                        : wg + (size_t)(row - ND - NTH) * KDIM;
    const float4 f = ((const float4*)src)[k4];
    v4h h = { (_Float16)f.x, (_Float16)f.y, (_Float16)f.z, (_Float16)f.w };
    ((v4h*)(o + (size_t)row * KDIM))[k4] = h;
}

// ---- stage one tile (ROWS x KC halves) from global into LDS (padded stride) ----
template <int ROWS>
__device__ __forceinline__ void hrl_copy_tile(const _Float16* __restrict__ g,
                                              _Float16* s, int tid) {
    constexpr int CHUNKS = ROWS * (KC / 8);      // 16B chunks
    #pragma unroll
    for (int t = 0; t < CHUNKS / 256; ++t) {
        const int c  = tid + t * 256;
        const int r  = c >> 3;
        const int k8 = (c & 7) * 8;
        const _Float16* gp = g + (size_t)r * KDIM + k8;
        _Float16* sp = s + r * LDSS + k8;
#if HRL_HAS_ASYNC
        __builtin_amdgcn_global_load_async_to_lds_b128(
            (glob_v4i*)gp, (lds_v4i*)sp, 0, 0);
#else
        *(v8h*)sp = *(const v8h*)gp;
#endif
    }
}

// ---------------- WMMA GEMM: C(32768x672) = A(32768x2048) * W^T ----------------
// A row-major f16; W stored (672 x 2048) row-major f16 (= B^T); C row-major f32.
// Block: 8 waves, 128x96 tile, LDS double-buffered K stages of 64.
union AFrag { v16h v; v8h h[2]; };

__global__ __launch_bounds__(256) void hrl_gemm_wmma(const _Float16* __restrict__ A,
                                                     const _Float16* __restrict__ W,
                                                     float* __restrict__ C) {
    __shared__ _Float16 sA[2][MB * LDSS];        // 2 * 18 KB
    __shared__ _Float16 sB[2][NB * LDSS];        // 2 * 13.5 KB

    const int tid  = threadIdx.x;
    const int lane = tid & 31;
    const int wid  = tid >> 5;
    const int l16  = lane & 15;
    const int hi   = lane >> 4;
    const int mBlk = (wid & 3) * 32;             // wave's 32-row group within block tile
    const int nBlk = (wid >> 2) * 48;            // wave's 48-col group within block tile

    const int m0 = blockIdx.x * MB;
    const int n0 = blockIdx.y * NB;

    const _Float16* gA = A + (size_t)m0 * KDIM;
    const _Float16* gB = W + (size_t)n0 * KDIM;

    v8f acc[2][3] = {};

    hrl_copy_tile<MB>(gA, sA[0], tid);
    hrl_copy_tile<NB>(gB, sB[0], tid);

    for (int kt = 0; kt < KDIM / KC; ++kt) {
        const int cur = kt & 1;
        hrl_wait_async();                        // own async fills of `cur` done
        __syncthreads();                         // whole tile visible; prev reads done
        if (kt + 1 < KDIM / KC) {                // overlap next-stage DMA with compute
            hrl_copy_tile<MB>(gA + (kt + 1) * KC, sA[cur ^ 1], tid);
            hrl_copy_tile<NB>(gB + (kt + 1) * KC, sB[cur ^ 1], tid);
        }
        const _Float16* a_base = sA[cur];
        const _Float16* b_base = sB[cur];
        #pragma unroll
        for (int kk = 0; kk < KC; kk += 32) {
            // B frags: lane = column, lanes 0-15 hold K kk..+7 & kk+16..+23; lanes 16-31 shifted by 8
            AFrag bf[3];
            #pragma unroll
            for (int ni = 0; ni < 3; ++ni) {
                const _Float16* bp = b_base + (nBlk + ni * 16 + l16) * LDSS + kk + hi * 8;
                bf[ni].h[0] = *(const v8h*)bp;
                bf[ni].h[1] = *(const v8h*)(bp + 16);
            }
            // A frags per ISA layout: lanes 0-15 K kk..+7 & kk+16..+23; lanes 16-31 shifted by 8
            #pragma unroll
            for (int mi = 0; mi < 2; ++mi) {
                AFrag af;
                const _Float16* ap = a_base + (mBlk + mi * 16 + l16) * LDSS + kk + hi * 8;
                af.h[0] = *(const v8h*)ap;
                af.h[1] = *(const v8h*)(ap + 16);
                #pragma unroll
                for (int ni = 0; ni < 3; ++ni)
                    acc[mi][ni] = __builtin_amdgcn_wmma_f32_16x16x32_f16(
                        false, af.v, false, bf[ni].v, (short)0, acc[mi][ni], false, false);
            }
        }
    }

    // C/D layout: VGPR i, lanes 0-15 -> (M=i, N=lane); lanes 16-31 -> (M=8+i, N=lane-16)
    #pragma unroll
    for (int mi = 0; mi < 2; ++mi) {
        #pragma unroll
        for (int ni = 0; ni < 3; ++ni) {
            float* c = C + (size_t)(m0 + mBlk + mi * 16 + hi * 8) * NTOT
                         + (n0 + nBlk + ni * 16 + l16);
            #pragma unroll
            for (int i = 0; i < 8; ++i) c[(size_t)i * NTOT] = acc[mi][ni][i];
        }
    }
}

// ---------------- Kuramoto dynamics, in-place on d_out ----------------
__device__ __forceinline__ float waveAllSum(float v) {
    #pragma unroll
    for (int off = 16; off > 0; off >>= 1) v += __shfl_xor(v, off, 32);
    return v;
}
__device__ __forceinline__ float pymod2pi(float x) {
    float r = fmodf(x, TWO_PI_F);
    return (r < 0.f) ? r + TWO_PI_F : r;
}

__global__ __launch_bounds__(256) void hrl_dynamics(float* __restrict__ Z,
                                                    const float* __restrict__ pdt_p,
                                                    const float* __restrict__ ptg_p) {
    const int lane = threadIdx.x & 31;
    const int row  = blockIdx.x * WPB + (threadIdx.x >> 5);
    float* p = Z + (size_t)row * NTOT;

    const float pac_dt = fminf(fmaxf(*pdt_p, 0.f), 1.f);
    const float pac_tg = fminf(fmaxf(*ptg_p, 0.f), 1.f);

    const float zd = p[lane];
    float dp = pymod2pi(zd);
    p[lane] = fmaxf(fabsf(zd), EPS_F);           // delta amplitude is invariant

    float tp[4], ta[4];
    #pragma unroll
    for (int j = 0; j < 4; ++j) {
        const float z = p[ND + j * 32 + lane];
        ta[j] = fmaxf(fabsf(z), EPS_F);
        tp[j] = pymod2pi(z);
    }
    float gp[16], ga[16];
    #pragma unroll
    for (int j = 0; j < 16; ++j) {
        const float z = p[ND + NTH + j * 32 + lane];
        ga[j] = fmaxf(fabsf(z), EPS_F);
        gp[j] = pymod2pi(z);
    }

    #pragma unroll 1
    for (int s = 0; s < 10; ++s) {
        const float cd = __cosf(dp), sd = __sinf(dp);
        const float Cd = waveAllSum(cd) * (1.f / 32.f);
        const float Sd = waveAllSum(sd) * (1.f / 32.f);

        float ct[4], st[4], sc = 0.f, ss = 0.f;
        #pragma unroll
        for (int j = 0; j < 4; ++j) {
            ct[j] = __cosf(tp[j]); st[j] = __sinf(tp[j]);
            sc += ct[j]; ss += st[j];
        }
        const float Ct = waveAllSum(sc) * (1.f / 128.f);
        const float St = waveAllSum(ss) * (1.f / 128.f);

        float cg[16], sg[16], gc = 0.f, gs = 0.f;
        #pragma unroll
        for (int j = 0; j < 16; ++j) {
            cg[j] = __cosf(gp[j]); sg[j] = __sinf(gp[j]);
            gc += cg[j]; gs += sg[j];
        }
        const float Cg = waveAllSum(gc) * (1.f / 512.f);
        const float Sg = waveAllSum(gs) * (1.f / 512.f);

        dp = pymod2pi(dp + DT_F * (TWO_PI_F * F_DELTA_F + KC_F * (Sd * cd - Cd * sd)));
        #pragma unroll
        for (int j = 0; j < 4; ++j)
            tp[j] = pymod2pi(tp[j] + DT_F * (TWO_PI_F * F_THETA_F + KC_F * (St * ct[j] - Ct * st[j])));
        #pragma unroll
        for (int j = 0; j < 16; ++j)
            gp[j] = pymod2pi(gp[j] + DT_F * (TWO_PI_F * F_GAMMA_F + KC_F * (Sg * cg[j] - Cg * sg[j])));

        const float rd = sqrtf(Cd * Cd + Sd * Sd) + EPS_F;
        const float rt = sqrtf(Ct * Ct + St * St) + EPS_F;
        const float fd = 1.f + DT_F * pac_dt * (Cd / rd);
        const float ft = 1.f + DT_F * pac_tg * (Ct / rt);
        #pragma unroll
        for (int j = 0; j < 4; ++j)  ta[j] = fmaxf(ta[j] * fd, EPS_F);
        #pragma unroll
        for (int j = 0; j < 16; ++j) ga[j] = fmaxf(ga[j] * ft, EPS_F);
    }

    #pragma unroll
    for (int j = 0; j < 4; ++j)  p[ND + j * 32 + lane] = ta[j];
    #pragma unroll
    for (int j = 0; j < 16; ++j) p[ND + NTH + j * 32 + lane] = ga[j];
}

extern "C" void kernel_launch(void* const* d_in, const int* in_sizes, int n_in,
                              void* d_out, int out_size, void* d_ws, size_t ws_size,
                              hipStream_t stream) {
    const float* x   = (const float*)d_in[0];
    const float* wd  = (const float*)d_in[1];
    const float* wt  = (const float*)d_in[2];
    const float* wg  = (const float*)d_in[3];
    const float* pdt = (const float*)d_in[4];
    const float* ptg = (const float*)d_in[5];
    float* out = (float*)d_out;

    _Float16* xh = (_Float16*)d_ws;                    // 32768*2048 f16 = 134 MB (L2-resident)
    _Float16* wh = xh + (size_t)B_ROWS * KDIM;         // 672*2048 f16  = 2.75 MB

    // 1) convert x and pack weights to f16
    hrl_cvt_x_f16<<<4096, 256, 0, stream>>>(x, xh, (B_ROWS * KDIM) / 4);
    {
        const int n4 = (NTOT * KDIM) / 4;
        hrl_cvt_w_f16<<<(n4 + 255) / 256, 256, 0, stream>>>(wd, wt, wg, wh, n4);
    }
    // 2) fused projection GEMM: 128x96 block tiles, async LDS double buffering, WMMA
    {
        dim3 grid(B_ROWS / MB, NTOT / NB);             // 256 x 7
        hrl_gemm_wmma<<<grid, 256, 0, stream>>>(xh, wh, out);
    }
    // 3) Kuramoto dynamics in-place on d_out (one wave32 per batch row)
    hrl_dynamics<<<B_ROWS / WPB, 256, 0, stream>>>(out, pdt, ptg);
}
